// Model_Net_11063835755139
// MI455X (gfx1250) — matrix operand order repaired and checked by reading, not compiled, not run on hardware
//
#include <hip/hip_runtime.h>
#include <hip/hip_bf16.h>
#include <stddef.h>

typedef __attribute__((ext_vector_type(16))) _Float16 v16h;
typedef __attribute__((ext_vector_type(8)))  float    v8f;

#define N_NODES 100000
#define N_EDGES 400000
#define NB 64

__device__ __forceinline__ float leakyf(float x) { return x >= 0.f ? x : 0.01f * x; }

// ---------------- elementwise / GCN helpers ----------------
__global__ void k_zero(float* p, int n) {
    int i = blockIdx.x * blockDim.x + threadIdx.x;
    if (i < n) p[i] = 0.f;
}
__global__ void k_fill1(float* p, int n) {
    int i = blockIdx.x * blockDim.x + threadIdx.x;
    if (i < n) p[i] = 1.f;
}
__global__ void k_relu(float* p, int n) {
    int i = blockIdx.x * blockDim.x + threadIdx.x;
    if (i < n) p[i] = fmaxf(p[i], 0.f);
}
__global__ void k_rsqrt(const float* a, float* b, int n) {
    int i = blockIdx.x * blockDim.x + threadIdx.x;
    if (i < n) b[i] = rsqrtf(a[i]);
}
__global__ void k_deg_accum(const int* __restrict__ dst, float* deg, int E) {
    int i = blockIdx.x * blockDim.x + threadIdx.x;
    if (i < E) atomicAdd(&deg[dst[i]], 1.f);
}
// out[node,f] = h[node,f]*dinv[node]^2 + bias[f]   (self-loop contribution + bias)
__global__ void k_agg_init(const float* __restrict__ h, const float* __restrict__ dinv,
                           const float* __restrict__ bias, float* __restrict__ out, int F) {
    int i = blockIdx.x * blockDim.x + threadIdx.x;
    int tot = N_NODES * F;
    if (i >= tot) return;
    int node = i / F, f = i - node * F;
    float di = dinv[node];
    out[i] = h[i] * di * di + bias[f];
}
__global__ void k_agg_edges(const int* __restrict__ src, const int* __restrict__ dst,
                            const float* __restrict__ h, const float* __restrict__ dinv,
                            float* __restrict__ out, int F) {
    int i = blockIdx.x * blockDim.x + threadIdx.x;
    int tot = N_EDGES * F;
    if (i >= tot) return;
    int e = i / F, f = i - e * F;
    int s = src[e], d = dst[e];
    atomicAdd(&out[(size_t)d * F + f], h[(size_t)s * F + f] * dinv[s] * dinv[d]);
}
__global__ void k_cnt_accum(const int* __restrict__ batch, float* cnt, int n) {
    int i = blockIdx.x * blockDim.x + threadIdx.x;
    if (i < n) atomicAdd(&cnt[batch[i]], 1.f);
}
__global__ void k_pool_accum(const int* __restrict__ batch, const float* __restrict__ x,
                             float* pooled, int F) {
    int i = blockIdx.x * blockDim.x + threadIdx.x;
    int tot = N_NODES * F;
    if (i >= tot) return;
    int node = i / F, f = i - node * F;
    atomicAdd(&pooled[(size_t)batch[node] * F + f], x[i]);
}
__global__ void k_pool_div(float* pooled, const float* __restrict__ cnt, int F) {
    int i = blockIdx.x * blockDim.x + threadIdx.x;
    int tot = NB * F;
    if (i >= tot) return;
    int b = i / F;
    pooled[i] /= fmaxf(cnt[b], 1.f);
}

// ---------------- f32 -> f16 staging (padded), weights transposed to [N,K] ----------------
__global__ void k_cvt_a(const float* __restrict__ src, _Float16* __restrict__ dst,
                        int M, int K, int Mp, int Kp) {
    int i = blockIdx.x * blockDim.x + threadIdx.x;
    int tot = Mp * Kp;
    if (i >= tot) return;
    int m = i / Kp, k = i - m * Kp;
    float v = (m < M && k < K) ? src[(size_t)m * K + k] : 0.f;
    dst[i] = (_Float16)v;
}
__global__ void k_cvt_bt(const float* __restrict__ W, _Float16* __restrict__ dst,
                         int K, int N, int Np, int Kp) {
    int i = blockIdx.x * blockDim.x + threadIdx.x;
    int tot = Np * Kp;
    if (i >= tot) return;
    int n = i / Kp, k = i - n * Kp;
    float v = (n < N && k < K) ? W[(size_t)k * N + n] : 0.f;
    dst[i] = (_Float16)v;
}

// ---------------- WMMA GEMM: C[M,N] = A @ B (+bias)(+relu) ----------------
// A row-major f16 [Mp,Kp]; Bt row-major f16 [Np64,Kp]. One wave -> 16x64 output tile
// (4 accumulators share one A fragment per K-step: 40 u32 loads per 4 WMMAs).
union FragU { v16h v; unsigned u[8]; };

__device__ __forceinline__ void store_tile(const v8f& acc, float* __restrict__ C,
                                           const float* __restrict__ bias,
                                           int M, int N, int act, int mb, int n) {
    if (n >= N) return;
    float bv = bias ? bias[n] : 0.f;
#pragma unroll
    for (int r = 0; r < 8; ++r) {
        int m = mb + r;
        if (m < M) {
            float v = acc[r] + bv;
            if (act) v = fmaxf(v, 0.f);
            C[(size_t)m * N + n] = v;
        }
    }
}

__global__ void k_wmma_gemm(const _Float16* __restrict__ A, const _Float16* __restrict__ Bt,
                            const float* __restrict__ bias, float* __restrict__ C,
                            int M, int N, int Kp, int Mp, int act) {
    int lane = threadIdx.x & 31;
    int wv   = threadIdx.x >> 5;
    int tM   = (blockIdx.x * 4 + wv) * 16;
    int tN0  = blockIdx.y * 64;
    if (tM >= Mp) return;

    const unsigned* arow  = (const unsigned*)(A + (size_t)(tM + (lane & 15)) * Kp);
    const unsigned* brow0 = (const unsigned*)(Bt + (size_t)(tN0 +  0 + (lane & 15)) * Kp);
    const unsigned* brow1 = (const unsigned*)(Bt + (size_t)(tN0 + 16 + (lane & 15)) * Kp);
    const unsigned* brow2 = (const unsigned*)(Bt + (size_t)(tN0 + 32 + (lane & 15)) * Kp);
    const unsigned* brow3 = (const unsigned*)(Bt + (size_t)(tN0 + 48 + (lane & 15)) * Kp);
    const int koffA = (lane & 16) ? 16 : 0;   // A: lanes 16-31 hold K=16..31 half
    const int koffB = (lane & 16) ? 16 : 0;   // B: lanes 16-31 hold K=16..31 half

    v8f acc0 = {}, acc1 = {}, acc2 = {}, acc3 = {};
    for (int kb = 0; kb < Kp; kb += 32) {
        if (kb + 32 < Kp)
            __builtin_prefetch(arow + ((kb + 32 + koffA) >> 1), 0, 0);
        FragU a, b0, b1, b2, b3;
#pragma unroll
        for (int i = 0; i < 8; ++i) {
            int ka = 2 * i + koffA;
            a.u[i] = arow[(kb + ka) >> 1];
        }
#pragma unroll
        for (int i = 0; i < 8; ++i) {
            int kbb = (kb + 2 * i + koffB) >> 1;
            b0.u[i] = brow0[kbb];
            b1.u[i] = brow1[kbb];
            b2.u[i] = brow2[kbb];
            b3.u[i] = brow3[kbb];
        }
        acc0 = __builtin_amdgcn_wmma_f32_16x16x32_f16(false, a.v, false, b0.v, (short)0, acc0, false, false);
        acc1 = __builtin_amdgcn_wmma_f32_16x16x32_f16(false, a.v, false, b1.v, (short)0, acc1, false, false);
        acc2 = __builtin_amdgcn_wmma_f32_16x16x32_f16(false, a.v, false, b2.v, (short)0, acc2, false, false);
        acc3 = __builtin_amdgcn_wmma_f32_16x16x32_f16(false, a.v, false, b3.v, (short)0, acc3, false, false);
    }
    int mb = tM + ((lane & 16) ? 8 : 0);
    int nn = lane & 15;
    store_tile(acc0, C, bias, M, N, act, mb, tN0 +  0 + nn);
    store_tile(acc1, C, bias, M, N, act, mb, tN0 + 16 + nn);
    store_tile(acc2, C, bias, M, N, act, mb, tN0 + 32 + nn);
    store_tile(acc3, C, bias, M, N, act, mb, tN0 + 48 + nn);
}

// ---------------- 3D branch ----------------
// site_3d [64,30,30,30,8] -> [64,8,30,30,30]
__global__ void k_site_tr(const float* __restrict__ in, float* __restrict__ out) {
    int i = blockIdx.x * blockDim.x + threadIdx.x;
    int tot = NB * 8 * 27000;
    if (i >= tot) return;
    int b = i / (8 * 27000);
    int r = i - b * 8 * 27000;
    int c = r / 27000;
    int sp = r - c * 27000;
    int z = sp / 900;
    int rr = sp - z * 900;
    int y = rr / 30;
    int x = rr - y * 30;
    out[i] = in[((size_t)((b * 30 + z) * 30 + y) * 30 + x) * 8 + c];
}
// one block per (b, oc, z): weights for oc cached in LDS
__global__ void k_conv3d(const float* __restrict__ in, const float* __restrict__ w,
                         const float* __restrict__ bias, float* __restrict__ out,
                         int IC, int OC, int D) {
    int OD = D - 2;
    int bid = blockIdx.x;
    int z  = bid % OD; int t = bid / OD;
    int oc = t % OC;   int b = t / OC;

    __shared__ float wsh[64 * 27];  // max IC = 64
    int nw = IC * 27;
    for (int i = threadIdx.x; i < nw; i += blockDim.x)
        wsh[i] = w[(size_t)oc * nw + i];
    __syncthreads();

    float bb = bias[oc];
    const float* ib = in + (size_t)b * IC * D * D * D;
    float* ob = out + (((size_t)(b * OC + oc) * OD + z) * OD) * OD;
    int npix = OD * OD;
    for (int p = threadIdx.x; p < npix; p += blockDim.x) {
        int y = p / OD, x = p - y * OD;
        float s = bb;
        for (int ic = 0; ic < IC; ++ic) {
            const float* ip = ib + (size_t)ic * D * D * D;
            const float* wp = wsh + ic * 27;
            for (int kz = 0; kz < 3; ++kz)
                for (int ky = 0; ky < 3; ++ky)
                    for (int kx = 0; kx < 3; ++kx)
                        s += ip[(size_t)(z + kz) * D * D + (y + ky) * D + (x + kx)] *
                             wp[kz * 9 + ky * 3 + kx];
        }
        ob[p] = leakyf(s);
    }
}
__global__ void k_pool3d(const float* __restrict__ in, float* __restrict__ out, int C, int D) {
    int OD = D / 2;
    int tot = NB * C * OD * OD * OD;
    int i = blockIdx.x * blockDim.x + threadIdx.x;
    if (i >= tot) return;
    int x = i % OD; int t = i / OD;
    int y = t % OD; t /= OD;
    int z = t % OD; t /= OD;
    int c = t % C; int b = t / C;
    const float* ip = in + (size_t)(b * C + c) * D * D * D;
    float m = -3.4e38f;
    for (int dz = 0; dz < 2; ++dz)
        for (int dy = 0; dy < 2; ++dy)
            for (int dx = 0; dx < 2; ++dx)
                m = fmaxf(m, ip[(size_t)(2 * z + dz) * D * D + (2 * y + dy) * D + (2 * x + dx)]);
    out[i] = m;
}

// ---------------- sequence branch ----------------
__global__ void k_embed(const int* __restrict__ tok, const float* __restrict__ Emb,
                        float* __restrict__ out) {
    int i = blockIdx.x * blockDim.x + threadIdx.x;
    int tot = NB * 128 * 994;
    if (i >= tot) return;
    int t = i % 994; int r = i / 994;
    int c = r % 128; int b = r / 128;
    out[i] = Emb[(size_t)tok[b * 994 + t] * 128 + c];
}
// one block per (b, oc): weights for oc cached in LDS
__global__ void k_conv1d(const float* __restrict__ in, const float* __restrict__ w,
                         const float* __restrict__ bias, float* __restrict__ out,
                         int IC, int OC, int L) {
    int OL = L + 2; // pad (3,3), kernel 5
    int bid = blockIdx.x;
    int oc = bid % OC; int b = bid / OC;

    __shared__ float wsh[128 * 5];  // max IC = 128
    int nw = IC * 5;
    for (int i = threadIdx.x; i < nw; i += blockDim.x)
        wsh[i] = w[(size_t)oc * nw + i];
    __syncthreads();

    float bb = bias[oc];
    const float* ib = in + (size_t)b * IC * L;
    float* ob = out + (size_t)(b * OC + oc) * OL;
    for (int t = threadIdx.x; t < OL; t += blockDim.x) {
        float s = bb;
        for (int ic = 0; ic < IC; ++ic) {
            const float* ip = ib + (size_t)ic * L;
            const float* wp = wsh + ic * 5;
            for (int j = 0; j < 5; ++j) {
                int p = t - 3 + j;
                if (p >= 0 && p < L) s += ip[p] * wp[j];
            }
        }
        ob[t] = leakyf(s);
    }
}
__global__ void k_pool1d(const float* __restrict__ in, float* __restrict__ out, int C, int L) {
    int OL = L / 2;
    int tot = NB * C * OL;
    int i = blockIdx.x * blockDim.x + threadIdx.x;
    if (i >= tot) return;
    int t = i % OL; int r = i / OL;
    int c = r % C; int b = r / C;
    const float* ip = in + (size_t)(b * C + c) * L;
    out[i] = fmaxf(ip[2 * t], ip[2 * t + 1]);
}

// ---------------- fusion head ----------------
__global__ void k_concat(const float* __restrict__ xg, const float* __restrict__ pv,
                         const float* __restrict__ sv, float* __restrict__ xc) {
    int i = blockIdx.x * blockDim.x + threadIdx.x;
    int tot = NB * 512;
    if (i >= tot) return;
    int b = i / 512, j = i - b * 512;
    float v;
    if (j < 256)      v = xg[b * 256 + j];
    else if (j < 384) v = pv[b * 128 + (j - 256)];
    else              v = sv[b * 128 + (j - 384)];
    xc[i] = v;
}
__global__ void k_final(const float* __restrict__ xc, const float* __restrict__ Wo,
                        const float* __restrict__ bo, float* __restrict__ out) {
    int b = blockIdx.x * blockDim.x + threadIdx.x;
    if (b >= NB) return;
    float s = bo[0];
    for (int k = 0; k < 512; ++k) s += xc[(size_t)b * 512 + k] * Wo[k];
    out[b] = s;
}

// ---------------- host-side helpers ----------------
static inline int cdiv(int a, int b) { return (a + b - 1) / b; }

static void run_gemm(const float* A, const float* W, const float* bias, float* C,
                     int M, int K, int N, int act,
                     _Float16* Abuf, _Float16* Bbuf, hipStream_t st) {
    int Mp   = (M + 15) & ~15;
    int Kp   = (K + 31) & ~31;
    int Np64 = (N + 63) & ~63;
    {
        int tot = Mp * Kp;
        k_cvt_a<<<cdiv(tot, 256), 256, 0, st>>>(A, Abuf, M, K, Mp, Kp);
    }
    {
        int tot = Np64 * Kp;
        k_cvt_bt<<<cdiv(tot, 256), 256, 0, st>>>(W, Bbuf, K, N, Np64, Kp);
    }
    dim3 g(cdiv(Mp / 16, 4), Np64 / 64);
    k_wmma_gemm<<<g, 128, 0, st>>>(Abuf, Bbuf, bias, C, M, N, Kp, Mp, act);
}

extern "C" void kernel_launch(void* const* d_in, const int* in_sizes, int n_in,
                              void* d_out, int out_size, void* d_ws, size_t ws_size,
                              hipStream_t stream) {
    (void)in_sizes; (void)n_in; (void)out_size; (void)ws_size;

    const float* mol_x  = (const float*)d_in[0];
    const int*   esrc   = (const int*)d_in[1];
    const int*   edst   = esrc + N_EDGES;
    const int*   batch  = (const int*)d_in[2];
    const float* site3d = (const float*)d_in[3];
    const int*   tokens = (const int*)d_in[4];
    const float *Wg1 = (const float*)d_in[5],  *bg1 = (const float*)d_in[6];
    const float *Wg2 = (const float*)d_in[7],  *bg2 = (const float*)d_in[8];
    const float *Wg3 = (const float*)d_in[9],  *bg3 = (const float*)d_in[10];
    const float *Wfg1 = (const float*)d_in[11], *bfg1 = (const float*)d_in[12];
    const float *Wfg2 = (const float*)d_in[13], *bfg2 = (const float*)d_in[14];
    const float *K1 = (const float*)d_in[15], *kb1 = (const float*)d_in[16];
    const float *K2 = (const float*)d_in[17], *kb2 = (const float*)d_in[18];
    const float *K3 = (const float*)d_in[19], *kb3 = (const float*)d_in[20];
    const float *Wp1 = (const float*)d_in[21], *bp1 = (const float*)d_in[22];
    const float *Wp2 = (const float*)d_in[23], *bp2 = (const float*)d_in[24];
    const float *Emb = (const float*)d_in[25];
    const float *C1 = (const float*)d_in[26], *cb1 = (const float*)d_in[27];
    const float *C2 = (const float*)d_in[28], *cb2 = (const float*)d_in[29];
    const float *C3 = (const float*)d_in[30], *cb3 = (const float*)d_in[31];
    const float *Ws1 = (const float*)d_in[32], *bs1 = (const float*)d_in[33];
    const float *Ws2 = (const float*)d_in[34], *bs2 = (const float*)d_in[35];
    const float *Wf1 = (const float*)d_in[36], *bf1 = (const float*)d_in[37];
    const float *Wf2 = (const float*)d_in[38], *bf2 = (const float*)d_in[39];
    const float *Wo  = (const float*)d_in[40], *bo  = (const float*)d_in[41];
    float* out = (float*)d_out;

    // -------- workspace layout (phase-aliased) --------
    char* ws = (char*)d_ws;
    const size_t MB = 1024ull * 1024ull;
    char* R0 = ws;                 // 192MB: GCN X0 | conv3d ping A
    char* R1 = ws + 192 * MB;      // 192MB: GCN X1 | site transposed + conv3d ping B
    char* R2 = ws + 384 * MB;      // 120MB: GCN gemm-out H | embed + conv1d pings
    char* R3 = ws + 504 * MB;      //  36MB: f16 A staging
    char* R4 = ws + 540 * MB;      //   4MB: f16 Bt staging
    char* R5 = ws + 544 * MB;      //   4MB: misc small buffers

    float* X0   = (float*)R0;
    float* X1   = (float*)R1;
    float* Hbuf = (float*)R2;
    _Float16* Abuf = (_Float16*)R3;
    _Float16* Bbuf = (_Float16*)R4;

    float* deg    = (float*)(R5 + 0x000000);
    float* dinv   = (float*)(R5 + 0x080000);
    float* pooled = (float*)(R5 + 0x100000);
    float* cnt    = (float*)(R5 + 0x120000);
    float* fcgt   = (float*)(R5 + 0x130000);  // [64,1024]
    float* xg     = (float*)(R5 + 0x180000);  // [64,256]
    float* pfc1   = (float*)(R5 + 0x1A0000);  // [64,128]
    float* pv     = (float*)(R5 + 0x1B0000);  // [64,128]
    float* sfc1   = (float*)(R5 + 0x1C0000);  // [64,256]
    float* sv     = (float*)(R5 + 0x1E0000);  // [64,128]
    float* xcat   = (float*)(R5 + 0x1F0000);  // [64,512]
    float* f1     = (float*)(R5 + 0x220000);  // [64,1024]
    float* f2     = (float*)(R5 + 0x270000);  // [64,512]

    // ================= GCN branch =================
    k_fill1<<<cdiv(N_NODES, 256), 256, 0, stream>>>(deg, N_NODES);
    k_deg_accum<<<cdiv(N_EDGES, 256), 256, 0, stream>>>(edst, deg, N_EDGES);
    k_rsqrt<<<cdiv(N_NODES, 256), 256, 0, stream>>>(deg, dinv, N_NODES);

    struct { const float* W; const float* b; int K; int N; } gl[3] = {
        {Wg1, bg1, 73, 73}, {Wg2, bg2, 73, 146}, {Wg3, bg3, 146, 292}};
    const float* xin = mol_x;
    float* xout = X0;
    for (int l = 0; l < 3; ++l) {
        int K = gl[l].K, F = gl[l].N;
        run_gemm(xin, gl[l].W, nullptr, Hbuf, N_NODES, K, F, 0, Abuf, Bbuf, stream);
        int tot = N_NODES * F;
        k_agg_init<<<cdiv(tot, 256), 256, 0, stream>>>(Hbuf, dinv, gl[l].b, xout, F);
        int etot = N_EDGES * F;
        k_agg_edges<<<cdiv(etot, 256), 256, 0, stream>>>(esrc, edst, Hbuf, dinv, xout, F);
        k_relu<<<cdiv(tot, 256), 256, 0, stream>>>(xout, tot);
        xin = xout;
        xout = (xout == X0) ? X1 : X0;
    }
    k_zero<<<cdiv(NB * 292, 256), 256, 0, stream>>>(pooled, NB * 292);
    k_zero<<<1, 64, 0, stream>>>(cnt, NB);
    k_cnt_accum<<<cdiv(N_NODES, 256), 256, 0, stream>>>(batch, cnt, N_NODES);
    k_pool_accum<<<cdiv(N_NODES * 292, 256), 256, 0, stream>>>(batch, xin, pooled, 292);
    k_pool_div<<<cdiv(NB * 292, 256), 256, 0, stream>>>(pooled, cnt, 292);
    run_gemm(pooled, Wfg1, bfg1, fcgt, NB, 292, 1024, 1, Abuf, Bbuf, stream);
    run_gemm(fcgt, Wfg2, bfg2, xg, NB, 1024, 256, 0, Abuf, Bbuf, stream);

    // ================= 3D CNN branch (GCN big buffers now dead) =================
    float* PT  = (float*)R1;                 // [64,8,30,30,30]
    float* CB1 = (float*)R0;                 // conv outputs (max 180MB)
    float* CB2 = (float*)(R1 + 64 * MB);     // pool outputs (max 22.5MB)
    k_site_tr<<<cdiv(NB * 8 * 27000, 256), 256, 0, stream>>>(site3d, PT);

    k_conv3d<<<NB * 32 * 28, 256, 0, stream>>>(PT, K1, kb1, CB1, 8, 32, 30);
    k_pool3d<<<cdiv(NB * 32 * 14 * 14 * 14, 256), 256, 0, stream>>>(CB1, CB2, 32, 28);
    k_conv3d<<<NB * 64 * 12, 256, 0, stream>>>(CB2, K2, kb2, CB1, 32, 64, 14);
    k_pool3d<<<cdiv(NB * 64 * 6 * 6 * 6, 256), 256, 0, stream>>>(CB1, CB2, 64, 12);
    k_conv3d<<<NB * 128 * 4, 256, 0, stream>>>(CB2, K3, kb3, CB1, 64, 128, 6);
    k_pool3d<<<cdiv(NB * 128 * 2 * 2 * 2, 256), 256, 0, stream>>>(CB1, CB2, 128, 4);
    // CB2 = [64,1024] contiguous
    run_gemm(CB2, Wp1, bp1, pfc1, NB, 1024, 128, 0, Abuf, Bbuf, stream);
    run_gemm(pfc1, Wp2, bp2, pv, NB, 128, 128, 0, Abuf, Bbuf, stream);

    // ================= sequence branch (H region now dead) =================
    float* EMBb = (float*)R2;                // [64,128,994] 32.6MB
    float* SB1  = (float*)(R2 + 40 * MB);    // max [64,64,996] 16.3MB
    float* SB2  = (float*)(R2 + 60 * MB);    // max [64,64,498] 8.2MB
    k_embed<<<cdiv(NB * 128 * 994, 256), 256, 0, stream>>>(tokens, Emb, EMBb);

    k_conv1d<<<NB * 64, 256, 0, stream>>>(EMBb, C1, cb1, SB1, 128, 64, 994);
    k_pool1d<<<cdiv(NB * 64 * 498, 256), 256, 0, stream>>>(SB1, SB2, 64, 996);
    k_conv1d<<<NB * 32, 256, 0, stream>>>(SB2, C2, cb2, SB1, 64, 32, 498);
    k_pool1d<<<cdiv(NB * 32 * 250, 256), 256, 0, stream>>>(SB1, SB2, 32, 500);
    k_conv1d<<<NB * 32, 256, 0, stream>>>(SB2, C3, cb3, SB1, 32, 32, 250);
    k_pool1d<<<cdiv(NB * 32 * 126, 256), 256, 0, stream>>>(SB1, SB2, 32, 252);
    // SB2 = [64,4032] contiguous
    run_gemm(SB2, Ws1, bs1, sfc1, NB, 4032, 256, 0, Abuf, Bbuf, stream);
    run_gemm(sfc1, Ws2, bs2, sv, NB, 256, 128, 0, Abuf, Bbuf, stream);

    // ================= fusion head =================
    k_concat<<<cdiv(NB * 512, 256), 256, 0, stream>>>(xg, pv, sv, xcat);
    run_gemm(xcat, Wf1, bf1, f1, NB, 512, 1024, 1, Abuf, Bbuf, stream);
    run_gemm(f1, Wf2, bf2, f2, NB, 1024, 512, 1, Abuf, Bbuf, stream);
    k_final<<<1, 64, 0, stream>>>(f2, Wo, bo, out);
}
